// CapsuleNet_19894288515150
// MI455X (gfx1250) — compile-verified, matrix operand order checked by reading, not run on hardware
//
#include <hip/hip_runtime.h>
#include <math.h>

// ---------- types ----------
typedef __attribute__((ext_vector_type(16))) __bf16 bf16x16;
typedef __attribute__((ext_vector_type(8)))  float  f32x8;
typedef __attribute__((ext_vector_type(2)))  float  f32x2;
typedef __attribute__((ext_vector_type(4)))  unsigned int u32x4;
typedef __attribute__((ext_vector_type(8)))  int    i32x8;
typedef __attribute__((ext_vector_type(4)))  int    i32x4;

__device__ __forceinline__ unsigned short f2bf(float f){
  unsigned u = __float_as_uint(f);
  u += 0x7fffu + ((u >> 16) & 1u);          // round-to-nearest-even
  return (unsigned short)(u >> 16);
}

union FragBF { uint4 q[2]; bf16x16 v; };

// One lane's 16 bf16 of a 16x16x32 bf16 WMMA fragment: 16B chunks at +0 / +16 elem.
// Works for both global and LDS pointers (address space inferred after inlining).
__device__ __forceinline__ bf16x16 ld_frag(const unsigned short* p){
  FragBF f;
  f.q[0] = *(const uint4*)(p);
  f.q[1] = *(const uint4*)(p + 16);
  return f.v;
}

// ---------- TDM: DMA a 64-row x 64-elem bf16 tile of Wt into LDS -------------
// D# per CDNA5 ISA 8.3-8.6. data_size=2B; pad 16B after every 128B row
// -> LDS row stride 144B (36 words: conflict-free, 16B aligned).
// tensor = Wt [256 rows x 20736 elems], row stride 20736.
// clang-23 toolchain: 6-arg builtin (g0, g1, g2, g3, extra i32x8, cpol).
__device__ __forceinline__ void tdm_load_tile(unsigned lds_addr, unsigned long long ga){
  u32x4 g0;
  g0[0] = 1u;                                              // count=1, user mode
  g0[1] = lds_addr;                                        // LDS byte address
  g0[2] = (unsigned)ga;                                    // global_addr[31:0]
  g0[3] = ((unsigned)(ga >> 32) & 0x01FFFFFFu) | 0x80000000u; // addr[56:32] | type=2
  i32x8 g1;
  g1[0] = (int)((1u<<16) | (1u<<20) | (4u<<22) | (3u<<25)); // 2B elems, pad_en, ivl=32dw, amt=4dw
  g1[1] = (int)0x51000000u;                                // tensor_dim0 = 20736 (lo16 @ [63:48])
  g1[2] = (int)0x01000000u;                                // dim0 hi=0 | tensor_dim1 = 256
  g1[3] = (int)0x00400000u;                                // tensor_dim1 hi=0 | tile_dim0 = 64
  g1[4] = 64;                                              // tile_dim1 = 64
  g1[5] = 20736;                                           // tensor_dim0_stride (lo32)
  g1[6] = 0;
  g1[7] = 0;
  i32x4 z4 = {0,0,0,0};                                    // groups 2/3: 2-D tile
  i32x8 z8 = {0,0,0,0,0,0,0,0};                            // unused trailing group
  __builtin_amdgcn_tensor_load_to_lds(g0, g1, z4, z4, z8, 0);
}

// ---------- conv1: [512,1,28,28] -> h1 bf16 NHWC [512,20,20,256], bias+ReLU ---
__global__ void k_conv1(const float* __restrict__ x, const float* __restrict__ w,
                        const float* __restrict__ bias, unsigned short* __restrict__ h1){
  __shared__ float img[9*28];
  int b  = blockIdx.x / 20;
  int oy = blockIdx.x % 20;
  int t  = threadIdx.x;
  if (t < 252) img[t] = x[b*784 + (oy + t/28)*28 + (t%28)];
  __syncthreads();
  float acc[20];
  #pragma unroll
  for (int i = 0; i < 20; ++i) acc[i] = 0.f;
  const float* wp = w + t*81;
  for (int ky = 0; ky < 9; ++ky){
    const float* ir = img + ky*28;
    #pragma unroll
    for (int kx = 0; kx < 9; ++kx){
      float wv = wp[ky*9 + kx];
      #pragma unroll
      for (int ox = 0; ox < 20; ++ox) acc[ox] += wv * ir[ox + kx];
    }
  }
  float bv = bias[t];
  unsigned short* o = h1 + (size_t)(b*20 + oy)*20*256 + t;
  #pragma unroll
  for (int ox = 0; ox < 20; ++ox){
    float v = acc[ox] + bv;
    v = v > 0.f ? v : 0.f;
    o[ox*256] = f2bf(v);
  }
}

// ---------- conv2 weight reorder: [O,C,9,9] f32 -> Wt bf16 [O, ky*9+kx, C] ----
__global__ void k_w2(const float* __restrict__ w, unsigned short* __restrict__ Wt){
  int g = blockIdx.x*256 + threadIdx.x;          // 256*81*256 = 5,308,416
  int o = g / 20736;
  int rem = g % 20736;
  int kk = rem >> 8;
  int c  = rem & 255;
  Wt[g] = f2bf(w[o*20736 + c*81 + kk]);
}

// ---------- conv2 implicit GEMM, bf16 WMMA, TDM-staged B tiles ----------------
// M=18432 (b,oy,ox), N=256 (oc), K=20736. Wave tile 32x32, block 8 waves
// (4M x 2N) -> 128x64, grid 576. B tile (64 cols x 64 k) DMA'd to LDS by TDM,
// double-buffered; A fragments stream from L2. Output h2 fp32 [b][oc][36].
__global__ void k_conv2(const unsigned short* __restrict__ h1,
                        const unsigned short* __restrict__ Wt,
                        const float* __restrict__ bias,
                        float* __restrict__ h2){
  __shared__ unsigned short bsm[2][64*72];       // 64 rows x (64 data + 8 pad) elems
  int t = threadIdx.x;
  int l = t & 31, wave = t >> 5;
  int lm = l & 15, h = l >> 4;
  int waveM = wave >> 1, waveN = wave & 1;
  int bm = blockIdx.x >> 2, bn = blockIdx.x & 3;
  int mbase = bm*128 + waveM*32;
  int nbB   = bn*64;                             // block's N base
  int nl    = waveN*32 + lm;                     // lane's local B row (+ j*16)

  size_t aRow[2];
  #pragma unroll
  for (int i = 0; i < 2; ++i){
    int m = mbase + i*16 + lm;
    int bb = m / 36, s = m - bb*36;
    int oy = s / 6,  ox = s - oy*6;
    aRow[i] = (size_t)((bb*20 + 2*oy)*20 + 2*ox) * 256;
  }

  unsigned long long gbase = (unsigned long long)(const void*)(Wt + (size_t)nbB*20736);
  unsigned ldsb[2];
  ldsb[0] = (unsigned)(size_t)&bsm[0][0];
  ldsb[1] = (unsigned)(size_t)&bsm[1][0];

  const f32x8 z = {0.f,0.f,0.f,0.f,0.f,0.f,0.f,0.f};
  f32x8 acc[2][2];
  acc[0][0] = z; acc[0][1] = z; acc[1][0] = z; acc[1][1] = z;

  if (wave == 0) tdm_load_tile(ldsb[0], gbase);  // prefetch ksq=0

  for (int ksq = 0; ksq < 324; ++ksq){           // 324 chunks of K=64
    int cur = ksq & 1;
    if (wave == 0){
      if (ksq + 1 < 324){
        tdm_load_tile(ldsb[cur ^ 1], gbase + (unsigned long long)(ksq + 1)*128);
        __builtin_amdgcn_s_wait_tensorcnt(1);    // ksq's DMA done, ksq+1 in flight
      } else {
        __builtin_amdgcn_s_wait_tensorcnt(0);
      }
    }
    __syncthreads();                              // publish B tile to all waves
    #pragma unroll
    for (int half = 0; half < 2; ++half){
      int ks = ksq*2 + half;                      // K-step of 32
      int kk = ks >> 3, cs = ks & 7;
      int ky = kk/9, kx = kk - ky*9;
      int offA = (ky*20 + kx)*256 + cs*32 + h*8;
      bf16x16 a0 = ld_frag(h1 + aRow[0] + offA);
      bf16x16 a1 = ld_frag(h1 + aRow[1] + offA);
      const unsigned short* bp = &bsm[cur][0] + half*32 + h*8;
      bf16x16 b0 = ld_frag(bp + (size_t)nl*72);
      bf16x16 b1 = ld_frag(bp + (size_t)(nl + 16)*72);
      acc[0][0] = __builtin_amdgcn_wmma_f32_16x16x32_bf16(false, a0, false, b0, (short)0, acc[0][0], false, false);
      acc[0][1] = __builtin_amdgcn_wmma_f32_16x16x32_bf16(false, a0, false, b1, (short)0, acc[0][1], false, false);
      acc[1][0] = __builtin_amdgcn_wmma_f32_16x16x32_bf16(false, a1, false, b0, (short)0, acc[1][0], false, false);
      acc[1][1] = __builtin_amdgcn_wmma_f32_16x16x32_bf16(false, a1, false, b1, (short)0, acc[1][1], false, false);
    }
    __syncthreads();                              // done reading buf before overwrite
  }

  // C/D layout: lane l holds n = nbB + nl + j*16; VGPR g -> m = mbase+i*16+h*8+g
  #pragma unroll
  for (int j = 0; j < 2; ++j){
    int n = nbB + nl + j*16;
    float bv = bias[n];
    #pragma unroll
    for (int i = 0; i < 2; ++i){
      #pragma unroll
      for (int g = 0; g < 8; ++g){
        int m = mbase + i*16 + h*8 + g;
        int bb = m / 36, s = m - bb*36;
        h2[((size_t)bb*256 + n)*36 + s] = acc[i][j][g] + bv;
      }
    }
  }
}

// ---------- squash over capsules: h2 viewed [B,1152,8] -> u fp32 --------------
__global__ void k_squash(const float* __restrict__ h2, float* __restrict__ u){
  int idx = blockIdx.x*256 + threadIdx.x;        // 589824 capsules exactly
  const float* p = h2 + (size_t)idx*8;
  float v[8], sq = 0.f;
  #pragma unroll
  for (int i = 0; i < 8; ++i){ v[i] = p[i]; sq += v[i]*v[i]; }
  float scale = sq / ((1.f + sq) * sqrtf(sq));
  float* q = u + (size_t)idx*8;
  #pragma unroll
  for (int i = 0; i < 8; ++i) q[i] = v[i]*scale;
}

// ---------- u_hat: per-route GEMM with f32 WMMA 16x16x4 -----------------------
__global__ void k_uhat(const float* __restrict__ u, const float* __restrict__ W,
                       float* __restrict__ uh){
  int r = blockIdx.x >> 2;
  int mchunk = blockIdx.x & 3;
  int t = threadIdx.x, l = t & 31, wave = t >> 5;
  int lm = l & 15, h = l >> 4;
  int mbase = (mchunk*8 + wave)*16;

  const float* up = u + ((size_t)(mbase + lm)*1152 + r)*8;
  f32x2 a0, a1;
  a0.x = up[2*h];     a0.y = up[2*h + 1];
  a1.x = up[4 + 2*h]; a1.y = up[4 + 2*h + 1];

  const float* wr = W + (size_t)r*1280;          // W[r, c, o, i] = wr[n*8 + i]
  for (int nt = 0; nt < 10; ++nt){
    int n = nt*16 + lm;
    const float* wn = wr + n*8;
    f32x2 b0, b1;
    b0.x = wn[2*h];     b0.y = wn[2*h + 1];
    b1.x = wn[4 + 2*h]; b1.y = wn[4 + 2*h + 1];
    f32x8 c = {0.f,0.f,0.f,0.f,0.f,0.f,0.f,0.f};
    c = __builtin_amdgcn_wmma_f32_16x16x4_f32(false, a0, false, b0, (short)0, c, false, false);
    c = __builtin_amdgcn_wmma_f32_16x16x4_f32(false, a1, false, b1, (short)0, c, false, false);
    #pragma unroll
    for (int g = 0; g < 8; ++g){
      int bb = mbase + h*8 + g;
      uh[((size_t)bb*1152 + r)*160 + n] = c[g];
    }
  }
}

// ---------- routing: softmax over routes (per capsule class) ------------------
__global__ void k_softmax(const float* __restrict__ bij, float* __restrict__ cij){
  int c = blockIdx.x, t = threadIdx.x;           // grid 10, block 128
  __shared__ float red[128];
  float m = -1e30f;
  for (int r = t; r < 1152; r += 128) m = fmaxf(m, bij[r*10 + c]);
  red[t] = m; __syncthreads();
  for (int s = 64; s; s >>= 1){ if (t < s) red[t] = fmaxf(red[t], red[t + s]); __syncthreads(); }
  float smax = red[0]; __syncthreads();
  float sum = 0.f;
  for (int r = t; r < 1152; r += 128){
    float e = expf(bij[r*10 + c] - smax);
    cij[r*10 + c] = e; sum += e;
  }
  red[t] = sum; __syncthreads();
  for (int s = 64; s; s >>= 1){ if (t < s) red[t] += red[t + s]; __syncthreads(); }
  float inv = 1.f / red[0];
  for (int r = t; r < 1152; r += 128) cij[r*10 + c] *= inv;
}

// ---------- routing: s_j = sum_r c_ij * u_hat, then squash -> v_j -------------
__global__ void k_route(const float* __restrict__ uh, const float* __restrict__ cij,
                        float* __restrict__ vj){
  int b = blockIdx.x / 10, c = blockIdx.x % 10;  // grid 5120, block 256
  int t = threadIdx.x, o = t & 15, grp = t >> 4;
  const float* up = uh + (size_t)b*184320 + c*16 + o;
  float acc = 0.f;
  for (int r = grp; r < 1152; r += 16) acc += cij[r*10 + c] * up[(size_t)r*160];
  __shared__ float red[256];
  red[t] = acc; __syncthreads();
  for (int s = 8; s; s >>= 1){ if (grp < s) red[t] += red[t + s*16]; __syncthreads(); }
  if (t < 16){
    float sq = 0.f;
    #pragma unroll
    for (int j = 0; j < 16; ++j){ float v = red[j]; sq += v*v; }
    float scale = sq / ((1.f + sq) * sqrtf(sq));
    vj[(b*10 + c)*16 + t] = red[t]*scale;
  }
}

// ---------- routing: b_ij += mean_b sum_o u_hat * v_j -------------------------
__global__ void k_agree(const float* __restrict__ uh, const float* __restrict__ vj,
                        float* __restrict__ bij){
  int r = blockIdx.x / 10, c = blockIdx.x % 10;  // grid 11520, block 256
  int t = threadIdx.x, o = t & 15, grp = t >> 4;
  float acc = 0.f;
  for (int b = grp; b < 512; b += 16)
    acc += uh[((size_t)b*1152 + r)*160 + c*16 + o] * vj[(b*10 + c)*16 + o];
  __shared__ float red[256];
  red[t] = acc; __syncthreads();
  for (int s = 128; s; s >>= 1){ if (t < s) red[t] += red[t + s]; __syncthreads(); }
  if (t == 0) bij[r*10 + c] += red[0] * (1.0f/512.0f);
}

// -----------------------------------------------------------------------------
extern "C" void kernel_launch(void* const* d_in, const int* in_sizes, int n_in,
                              void* d_out, int out_size, void* d_ws, size_t ws_size,
                              hipStream_t stream){
  const float* x   = (const float*)d_in[0];
  const float* c1w = (const float*)d_in[1];
  const float* c1b = (const float*)d_in[2];
  const float* c2w = (const float*)d_in[3];
  const float* c2b = (const float*)d_in[4];
  const float* Wr  = (const float*)d_in[5];
  float* vj = (float*)d_out;                     // [512,10,16] fp32

  char* ws = (char*)d_ws;
  size_t off = 0;
  auto take = [&](size_t bytes) -> void* {
    void* p = ws + off;
    off += (bytes + 255) & ~(size_t)255;
    return p;
  };
  // ~531 MB total workspace
  unsigned short* h1 = (unsigned short*)take((size_t)512*400*256*2);   // bf16 NHWC
  unsigned short* Wt = (unsigned short*)take((size_t)256*20736*2);     // bf16 [O,kk,C]
  float* h2  = (float*)take((size_t)512*9216*4);                       // conv2 out
  float* u   = (float*)take((size_t)512*9216*4);                       // squashed caps
  float* uh  = (float*)take((size_t)512*1152*160*4);                   // u_hat
  float* bij = (float*)take((size_t)11520*4);
  float* cij = (float*)take((size_t)11520*4);

  k_conv1 <<<10240, 256, 0, stream>>>(x, c1w, c1b, h1);
  k_w2    <<<20736, 256, 0, stream>>>(c2w, Wt);
  k_conv2 <<<576,   256, 0, stream>>>(h1, Wt, c2b, h2);
  k_squash<<<2304,  256, 0, stream>>>(h2, u);
  k_uhat  <<<4608,  256, 0, stream>>>(u, Wr, uh);

  (void)hipMemsetAsync(bij, 0, (size_t)11520*4, stream);
  for (int it = 0; it < 3; ++it){
    k_softmax<<<10,    128, 0, stream>>>(bij, cij);
    k_route  <<<5120,  256, 0, stream>>>(uh, cij, vj);
    if (it < 2) k_agree<<<11520, 256, 0, stream>>>(uh, vj, bij);
  }
}
